// WindowAttention_11553462026660
// MI455X (gfx1250) — matrix operand order repaired and checked by reading, not compiled, not run on hardware
//
#include <hip/hip_runtime.h>
#include <hip/hip_bf16.h>

#define WIN_NTOK 16
#define WIN_DIM 128
#define WIN_SCALE 0.17677669529663687f /* 32^-0.5 */

typedef __attribute__((ext_vector_type(16))) __bf16        v16bf;
typedef __attribute__((ext_vector_type(8)))  float         v8f;
typedef __attribute__((ext_vector_type(8)))  unsigned int  v8u;

static __device__ __forceinline__ unsigned short f2bf(float f) {
    unsigned int u = __builtin_bit_cast(unsigned int, f);
    u += 0x7FFFu + ((u >> 16) & 1u);            // round-to-nearest-even
    return (unsigned short)(u >> 16);
}

static __device__ __forceinline__ v16bf to16(uint4 lo, uint4 hi) {
    v8u r;
    r[0] = lo.x; r[1] = lo.y; r[2] = lo.z; r[3] = lo.w;
    r[4] = hi.x; r[5] = hi.y; r[6] = hi.z; r[7] = hi.w;
    return __builtin_bit_cast(v16bf, r);
}

static __device__ __forceinline__ float rowmax16(float v) {
#pragma unroll
    for (int m = 1; m < 16; m <<= 1) v = fmaxf(v, __shfl_xor(v, m, 32));
    return v;
}
static __device__ __forceinline__ float rowsum16(float v) {
#pragma unroll
    for (int m = 1; m < 16; m <<= 1) v += __shfl_xor(v, m, 32);
    return v;
}

// ---------------------------------------------------------------------------
// Prep: swizzle weights into WMMA B-fragment layout (bf16) + bake rel-pos bias.
// B-fragment (32x16 K x N, bf16): lane l holds column n=l&15, K = 16*(l>>4)+j,
// j=0..15 packed ascending -> 32 contiguous bytes per lane per fragment (1 KB).
// ---------------------------------------------------------------------------
__global__ void winattn_prep(const float* __restrict__ w_qkv,
                             const float* __restrict__ w_proj,
                             const float* __restrict__ bias_table,
                             unsigned short* __restrict__ wsQ,   // 24*4 frags
                             unsigned short* __restrict__ wsP,   // 8*4  frags
                             float* __restrict__ wsB) {          // [4][16][16]
    const int blk  = blockIdx.x;
    const int lane = threadIdx.x;   // 32 threads
    if (blk < 96) {                 // w_qkv fragments: frag = nt*4 + kb
        const int nt = blk >> 2, kb = blk & 3;
        const int col   = nt * 16 + (lane & 15);
        const int kbase = kb * 32 + (lane >> 4) * 16;
        unsigned short* dst = wsQ + blk * 512 + lane * 16;
#pragma unroll
        for (int j = 0; j < 16; ++j)
            dst[j] = f2bf(w_qkv[(kbase + j) * 384 + col]);
    } else if (blk < 128) {         // w_proj fragments
        const int f  = blk - 96;
        const int nt = f >> 2, kb = f & 3;
        const int col   = nt * 16 + (lane & 15);
        const int kbase = kb * 32 + (lane >> 4) * 16;
        unsigned short* dst = wsP + f * 512 + lane * 16;
#pragma unroll
        for (int j = 0; j < 16; ++j)
            dst[j] = f2bf(w_proj[(kbase + j) * 128 + col]);
    } else {                        // relative position bias [h][t1][t2]
        for (int t = lane; t < 256; t += 32) {
            const int t1 = t >> 4, t2 = t & 15;
            const int r0 = (t1 >> 2) - (t2 >> 2) + 3;
            const int r1 = (t1 & 3) - (t2 & 3) + 3;
            const int idx = r0 * 7 + r1;
#pragma unroll
            for (int h = 0; h < 4; ++h)
                wsB[h * 256 + t] = bias_table[idx * 4 + h];
        }
    }
}

// ---------------------------------------------------------------------------
// Main: one wave32 per window, 8 waves per block.
// ---------------------------------------------------------------------------
__global__ void __launch_bounds__(256)
winattn_main(const float* __restrict__ x,       // [B,16,128]
             const float* __restrict__ mask,    // [256,16,16]
             const float* __restrict__ b_qkv,   // [384]
             const float* __restrict__ b_proj,  // [128]
             const unsigned short* __restrict__ wsQ,
             const unsigned short* __restrict__ wsP,
             const float* __restrict__ wsB,
             float* __restrict__ out) {         // [B,16,128]
    // per-wave LDS scratch (transpose staging)
    __shared__ __align__(16) unsigned short sQ[8][16 * 32];   // q[h][n][d]
    __shared__ __align__(16) unsigned short sK[8][16 * 32];   // k[h][n][d]
    __shared__ __align__(16) unsigned short sV[8][32 * 16];   // v^T[d][n]
    __shared__ __align__(16) unsigned short sA[8][16 * 16];   // attn[n][m]
    __shared__ __align__(16) unsigned short sO[8][16 * 128];  // pre-proj out

    const int wave = threadIdx.x >> 5;
    const int lane = threadIdx.x & 31;
    const int b    = blockIdx.x * 8 + wave;
    const int col  = lane & 15;
    const int half = lane >> 4;

    // ---- load x as WMMA A fragments (row m = col, bf16 convert in regs) ----
    const float* xw = x + (size_t)b * (16 * 128) + col * 128;
    v16bf axf[4];
#pragma unroll
    for (int kb = 0; kb < 4; ++kb) {
        const int k0 = kb * 32 + 16 * half;
        float4 f0 = *(const float4*)(xw + k0);
        float4 f1 = *(const float4*)(xw + k0 + 4);
        float4 f2 = *(const float4*)(xw + k0 + 8);
        float4 f3 = *(const float4*)(xw + k0 + 12);
        v8u p;
        p[0] = f2bf(f0.x) | ((unsigned)f2bf(f0.y) << 16);
        p[1] = f2bf(f0.z) | ((unsigned)f2bf(f0.w) << 16);
        p[2] = f2bf(f1.x) | ((unsigned)f2bf(f1.y) << 16);
        p[3] = f2bf(f1.z) | ((unsigned)f2bf(f1.w) << 16);
        p[4] = f2bf(f2.x) | ((unsigned)f2bf(f2.y) << 16);
        p[5] = f2bf(f2.z) | ((unsigned)f2bf(f2.w) << 16);
        p[6] = f2bf(f3.x) | ((unsigned)f2bf(f3.y) << 16);
        p[7] = f2bf(f3.z) | ((unsigned)f2bf(f3.w) << 16);
        axf[kb] = __builtin_bit_cast(v16bf, p);
    }

    // ---- window mask rows for this lane (m = r + 8*half, key = col) ----
    float mreg[8];
    {
        const float* mp = mask + (size_t)(b & 255) * 256 + (8 * half) * 16 + col;
#pragma unroll
        for (int r = 0; r < 8; ++r) mreg[r] = mp[r * 16];
    }

    const uint4 zero4 = {0u, 0u, 0u, 0u};

    for (int h = 0; h < 4; ++h) {
        // ------------- QKV GEMM for this head: 6 feature tiles -------------
        const int nts[6] = {2 * h, 2 * h + 1, 8 + 2 * h, 8 + 2 * h + 1,
                            16 + 2 * h, 16 + 2 * h + 1};
#pragma unroll
        for (int t = 0; t < 6; ++t) {
            const int nt = nts[t];
            v8f acc = {};
#pragma unroll
            for (int kb = 0; kb < 4; ++kb) {
                const uint4* bp = (const uint4*)(wsQ + (nt * 4 + kb) * 512 + lane * 16);
                v16bf bfrag = to16(bp[0], bp[1]);
                acc = __builtin_amdgcn_wmma_f32_16x16x32_bf16(
                    false, axf[kb], false, bfrag, (short)0, acc, false, false);
            }
            const float bias = b_qkv[nt * 16 + col];
            const int dh = (nt & 1) * 16;       // d offset within head
#pragma unroll
            for (int r = 0; r < 8; ++r) {
                float v = acc[r] + bias;
                if (t < 2) v *= WIN_SCALE;      // scale q
                const int n = r + 8 * half;
                const int d = dh + col;
                const unsigned short bv = f2bf(v);
                if (t < 2)      sQ[wave][n * 32 + d] = bv;
                else if (t < 4) sK[wave][n * 32 + d] = bv;
                else            sV[wave][d * 16 + n] = bv;  // transposed
            }
        }

        // ------------- attn = q @ k^T  (one WMMA, K = head_dim = 32) ------
        const uint4* qa0 = (const uint4*)&sQ[wave][col * 32 + 16 * half];
        const uint4* qa1 = (const uint4*)&sQ[wave][col * 32 + 16 * half + 8];
        v16bf qfrag = to16(qa0[0], qa1[0]);
        const uint4* kbp = (const uint4*)&sK[wave][col * 32 + 16 * half];
        v16bf kfrag = to16(kbp[0], kbp[1]);
        v8f att = {};
        att = __builtin_amdgcn_wmma_f32_16x16x32_bf16(
            false, qfrag, false, kfrag, (short)0, att, false, false);

        // ------------- bias + mask + double softmax over keys -------------
        const float* bpt = wsB + h * 256 + (8 * half) * 16 + col;
        float s[8];
#pragma unroll
        for (int r = 0; r < 8; ++r) s[r] = att[r] + bpt[r * 16] + mreg[r];
#pragma unroll
        for (int r = 0; r < 8; ++r) {
            float v  = s[r];
            float mx = rowmax16(v);
            float e  = __expf(v - mx);
            float p  = e / rowsum16(e);         // softmax #1
            float e2 = __expf(p);               // softmax #2 (p in [0,1])
            s[r] = e2 / rowsum16(e2);
        }
#pragma unroll
        for (int r = 0; r < 8; ++r)
            sA[wave][(r + 8 * half) * 16 + col] = f2bf(s[r]);

        // ------------- out_h = attn @ v  (K=16 zero-padded to 32) ---------
        uint4 alo = *(const uint4*)&sA[wave][col * 16];
        v16bf afrag = to16(alo, *(const uint4*)&sA[wave][col * 16 + 8]);
        const unsigned int kmask = half ? 0u : 0xFFFFFFFFu;
#pragma unroll
        for (int t = 0; t < 2; ++t) {
            uint4 v0 = *(const uint4*)&sV[wave][(16 * t + col) * 16];
            uint4 v1 = *(const uint4*)&sV[wave][(16 * t + col) * 16 + 8];
            v0.x &= kmask; v0.y &= kmask; v0.z &= kmask; v0.w &= kmask;
            v1.x &= kmask; v1.y &= kmask; v1.z &= kmask; v1.w &= kmask;
            v16bf vfrag = to16(v0, v1);
            v8f oacc = {};
            oacc = __builtin_amdgcn_wmma_f32_16x16x32_bf16(
                false, afrag, false, vfrag, (short)0, oacc, false, false);
#pragma unroll
            for (int r = 0; r < 8; ++r)
                sO[wave][(r + 8 * half) * 128 + h * 32 + 16 * t + col] = f2bf(oacc[r]);
        }
    }

    // ---------------- final projection: out @ w_proj + b_proj -------------
    v16bf ofrag[4];
#pragma unroll
    for (int kb = 0; kb < 4; ++kb) {
        const uint4* p0 = (const uint4*)&sO[wave][col * 128 + kb * 32 + 16 * half];
        const uint4* p1 = (const uint4*)&sO[wave][col * 128 + kb * 32 + 16 * half + 8];
        ofrag[kb] = to16(p0[0], p1[0]);
    }
    float* op = out + (size_t)b * (16 * 128);
#pragma unroll
    for (int nt = 0; nt < 8; ++nt) {
        v8f acc = {};
#pragma unroll
        for (int kb = 0; kb < 4; ++kb) {
            const uint4* bp = (const uint4*)(wsP + (nt * 4 + kb) * 512 + lane * 16);
            v16bf bfrag = to16(bp[0], bp[1]);
            acc = __builtin_amdgcn_wmma_f32_16x16x32_bf16(
                false, ofrag[kb], false, bfrag, (short)0, acc, false, false);
        }
        const float bias = b_proj[nt * 16 + col];
#pragma unroll
        for (int r = 0; r < 8; ++r)
            op[(r + 8 * half) * 128 + nt * 16 + col] = acc[r] + bias;
    }
}

extern "C" void kernel_launch(void* const* d_in, const int* in_sizes, int n_in,
                              void* d_out, int out_size, void* d_ws, size_t ws_size,
                              hipStream_t stream) {
    const float* x      = (const float*)d_in[0];
    const float* mask   = (const float*)d_in[1];
    const float* w_qkv  = (const float*)d_in[2];
    const float* b_qkv  = (const float*)d_in[3];
    const float* w_proj = (const float*)d_in[4];
    const float* b_proj = (const float*)d_in[5];
    const float* btab   = (const float*)d_in[6];
    float* out = (float*)d_out;

    unsigned short* wsQ = (unsigned short*)d_ws;   // 96 KB: qkv B-fragments
    unsigned short* wsP = wsQ + 96 * 512;          // 32 KB: proj B-fragments
    float*          wsB = (float*)(wsP + 32 * 512);// 4 KB : rel-pos bias

    winattn_prep<<<129, 32, 0, stream>>>(w_qkv, w_proj, btab, wsQ, wsP, wsB);

    const int B_ = in_sizes[0] / (WIN_NTOK * WIN_DIM);   // 16384 windows
    winattn_main<<<B_ / 8, 256, 0, stream>>>(x, mask, b_qkv, b_proj,
                                             wsQ, wsP, wsB, out);
}
